// BiMambaEncoder_58248346468741
// MI455X (gfx1250) — compile-verified, matrix operand order checked by reading, not run on hardware
//
#include <hip/hip_runtime.h>
#include <hip/hip_bf16.h>

// ---------------------------------------------------------------------------
// BiMamba encoder for MI455X (gfx1250, wave32, WMMA).
// GEMMs use v_wmma_f32_16x16x32_bf16 (bf16 A/B, f32 accum). Scan stays f32.
// ---------------------------------------------------------------------------

#define D_MODEL 256
#define D_INNER 512
#define D_STATE 16
#define DT_RANK 16
#define N_LAYERS 4
#define BATCH 16
#define SEQ 1024
#define MTOK (BATCH * SEQ)   // 16384 tokens

typedef __bf16 bf16_t;
typedef __attribute__((ext_vector_type(16))) __bf16 v16bf;
typedef __attribute__((ext_vector_type(8)))  __bf16 v8bf;
typedef __attribute__((ext_vector_type(8)))  float  v8f;

// ---------------------------------------------------------------------------
// f32 -> bf16 weight conversion
// ---------------------------------------------------------------------------
__global__ void cvt_bf16_kernel(const float* __restrict__ in, bf16_t* __restrict__ out, int n) {
    int i = blockIdx.x * 256 + threadIdx.x;
    if (i < n) out[i] = (bf16_t)in[i];
}

// f32 (rows x in_cols) -> bf16 (rows x out_cols), zero-padded K tail
__global__ void cvt_bf16_pad_kernel(const float* __restrict__ in, bf16_t* __restrict__ out,
                                    int rows, int in_cols, int out_cols) {
    int i = blockIdx.x * 256 + threadIdx.x;
    if (i >= rows * out_cols) return;
    int r = i / out_cols;
    int c = i - r * out_cols;
    out[i] = (c < in_cols) ? (bf16_t)in[r * in_cols + c] : (bf16_t)0.0f;
}

// ---------------------------------------------------------------------------
// WMMA fragment loaders (layouts per CDNA5 ISA 7.12.2, bf16 A 16x32 / B 32x16)
// lane l: row/col = l%16 ; kbase = (l>=16) ? 8 : 0
// element e<8 : K = k0 + kbase + e ; e>=8 : K = k0 + 16 + kbase + (e-8)
// Per lane: two contiguous runs of 8 -> fully vectorized b128 loads, no guards.
// ---------------------------------------------------------------------------
__device__ __forceinline__ v16bf load_a_frag_f32(const float* __restrict__ A, int lda,
                                                 int m0, int k0, int lane) {
    const float* p = A + (long)(m0 + (lane & 15)) * lda + k0 + ((lane & 16) ? 8 : 0);
    float4 r0 = *(const float4*)(p);
    float4 r1 = *(const float4*)(p + 4);
    float4 r2 = *(const float4*)(p + 16);
    float4 r3 = *(const float4*)(p + 20);
    v16bf a;
    a[0]  = (bf16_t)r0.x; a[1]  = (bf16_t)r0.y; a[2]  = (bf16_t)r0.z; a[3]  = (bf16_t)r0.w;
    a[4]  = (bf16_t)r1.x; a[5]  = (bf16_t)r1.y; a[6]  = (bf16_t)r1.z; a[7]  = (bf16_t)r1.w;
    a[8]  = (bf16_t)r2.x; a[9]  = (bf16_t)r2.y; a[10] = (bf16_t)r2.z; a[11] = (bf16_t)r2.w;
    a[12] = (bf16_t)r3.x; a[13] = (bf16_t)r3.y; a[14] = (bf16_t)r3.z; a[15] = (bf16_t)r3.w;
    return a;
}

__device__ __forceinline__ v16bf load_b_frag_bf16(const bf16_t* __restrict__ W, int ldw,
                                                  int n0, int k0, int lane) {
    const bf16_t* p = W + (long)(n0 + (lane & 15)) * ldw + k0 + ((lane & 16) ? 8 : 0);
    v8bf lo = *(const v8bf*)(p);       // 16B
    v8bf hi = *(const v8bf*)(p + 16);  // 16B
    v16bf b;
#pragma unroll
    for (int e = 0; e < 8; ++e) { b[e] = lo[e]; b[8 + e] = hi[e]; }
    return b;
}

__device__ __forceinline__ float softplus_f(float v) {
    return (v > 20.0f) ? v : log1pf(__expf(v));
}

// ---------------------------------------------------------------------------
// WMMA GEMM:  C(MxN) = A(MxK, f32) @ W(NxK, bf16)^T      (K % 32 == 0)
// block = 256 threads = 8 waves; wave -> (16*MT) rows x (16*NT) cols
// grid = (M/(8*16*MT), N/(16*NT))
// MODE (compile-time): 0 = plain, 1 = +bias[n], 2 = softplus(+bias[n]),
//                      3 = accumulate (+C)
// ---------------------------------------------------------------------------
template <int MT, int NT, int MODE>
__global__ void wmma_gemm_kernel(const float* __restrict__ A, int lda,
                                 const bf16_t* __restrict__ W, int ldw,
                                 float* __restrict__ C, int ldc,
                                 int M, int N, int K,
                                 const float* __restrict__ bias) {
    const int lane = threadIdx.x & 31;
    const int wave = threadIdx.x >> 5;
    const int m0 = (blockIdx.x * 8 + wave) * (16 * MT);
    const int n0 = blockIdx.y * (16 * NT);

    v8f acc[MT][NT];
#pragma unroll
    for (int mt = 0; mt < MT; ++mt)
#pragma unroll
        for (int t = 0; t < NT; ++t)
            acc[mt][t] = (v8f){0.f, 0.f, 0.f, 0.f, 0.f, 0.f, 0.f, 0.f};

    for (int k0 = 0; k0 < K; k0 += 32) {
        // speculative prefetch of next A K-slab (dropped silently if OOB)
        __builtin_prefetch(A + (long)(m0 + (lane & 15)) * lda + k0 + 32, 0, 0);
        v16bf a[MT];
#pragma unroll
        for (int mt = 0; mt < MT; ++mt)
            a[mt] = load_a_frag_f32(A, lda, m0 + 16 * mt, k0, lane);
#pragma unroll
        for (int t = 0; t < NT; ++t) {
            v16bf b = load_b_frag_bf16(W, ldw, n0 + 16 * t, k0, lane);
#pragma unroll
            for (int mt = 0; mt < MT; ++mt)
                acc[mt][t] = __builtin_amdgcn_wmma_f32_16x16x32_bf16(
                    false, a[mt], false, b, (short)0, acc[mt][t], false, false);
        }
    }

    // store: lane l -> N = n0 + l%16 ; vgpr r -> M = m0 + r + (l>=16 ? 8 : 0)
    const int nlane = lane & 15;
    const int mshift = (lane & 16) ? 8 : 0;
#pragma unroll
    for (int mt = 0; mt < MT; ++mt) {
        int mbase = m0 + 16 * mt + mshift;
#pragma unroll
        for (int t = 0; t < NT; ++t) {
            int n = n0 + 16 * t + nlane;
            float bval = (MODE == 1 || MODE == 2) ? bias[n] : 0.0f;
#pragma unroll
            for (int r = 0; r < 8; ++r) {
                int m = mbase + r;
                float v = acc[mt][t][r] + bval;
                if (MODE == 2) v = softplus_f(v);
                float* p = C + (long)m * ldc + n;
                if (MODE == 3) v += *p;
                *p = v;
            }
        }
    }
}

// ---------------------------------------------------------------------------
// Embedder: one block (256 threads) per token. cat(136) in LDS, 136->256
// matvec, blockwide LayerNorm. Writes both chain-init buffers.
// ---------------------------------------------------------------------------
__global__ void embed_kernel(const float* __restrict__ X,
                             const float* __restrict__ proto_emb,
                             const float* __restrict__ flags_emb,
                             const float* __restrict__ dir_emb,
                             const float* __restrict__ loglen_w,
                             const float* __restrict__ loglen_b,
                             const float* __restrict__ iat_w,
                             const float* __restrict__ iat_b,
                             const float* __restrict__ fusion_w,
                             const float* __restrict__ fusion_b,
                             const float* __restrict__ ln_g,
                             const float* __restrict__ ln_b,
                             float* __restrict__ cur_f,
                             float* __restrict__ cur_r) {
    __shared__ float cat[136];
    __shared__ float red[256];
    const int tok = blockIdx.x;
    const int tid = threadIdx.x;
    const float* xr = X + (long)tok * 5;

    if (tid < 136) {
        float v;
        if (tid < 32) {
            int p = (int)xr[0]; p = p < 0 ? 0 : (p > 255 ? 255 : p);
            v = proto_emb[p * 32 + tid];
        } else if (tid < 64) {
            int j = tid - 32;
            v = xr[1] * loglen_w[j] + loglen_b[j];
        } else if (tid < 96) {
            int f = (int)xr[2]; f = f < 0 ? 0 : (f > 63 ? 63 : f);
            v = flags_emb[f * 32 + (tid - 64)];
        } else if (tid < 128) {
            int j = tid - 96;
            v = xr[3] * iat_w[j] + iat_b[j];
        } else {
            int d = (int)xr[4]; d = d < 0 ? 0 : (d > 1 ? 1 : d);
            v = dir_emb[d * 8 + (tid - 128)];
        }
        cat[tid] = v;
    }
    __syncthreads();

    float h = fusion_b[tid];
    const float* wrow = fusion_w + (long)tid * 136;
#pragma unroll 8
    for (int j = 0; j < 136; ++j) h += cat[j] * wrow[j];

    // mean
    red[tid] = h;
    __syncthreads();
    for (int off = 128; off > 0; off >>= 1) {
        if (tid < off) red[tid] += red[tid + off];
        __syncthreads();
    }
    float mean = red[0] * (1.0f / 256.0f);
    __syncthreads();
    // variance
    float dv = h - mean;
    red[tid] = dv * dv;
    __syncthreads();
    for (int off = 128; off > 0; off >>= 1) {
        if (tid < off) red[tid] += red[tid + off];
        __syncthreads();
    }
    float var = red[0] * (1.0f / 256.0f);
    float o = dv * rsqrtf(var + 1e-5f) * ln_g[tid] + ln_b[tid];
    cur_f[(long)tok * D_MODEL + tid] = o;
    cur_r[(long)tok * D_MODEL + tid] = o;
}

// ---------------------------------------------------------------------------
// Depthwise causal conv (width 4) + SiLU.  dir=0: taps t-3..t (fwd chain);
// dir=1: taps t..t+3 with reversed weights (rev chain in natural order).
// Reads x-half of xz (row pitch 2*D_INNER), writes xi (MxD_INNER).
// ---------------------------------------------------------------------------
__global__ void conv_silu_kernel(const float* __restrict__ xz,
                                 const float* __restrict__ conv_w,
                                 const float* __restrict__ conv_b,
                                 float* __restrict__ xi, int dir) {
    long idx = (long)blockIdx.x * 256 + threadIdx.x; // over MTOK*D_INNER
    int c = (int)(idx & (D_INNER - 1));
    long bt = idx >> 9;
    int t = (int)(bt & (SEQ - 1));
    int b = (int)(bt >> 10);
    float acc = conv_b[c];
#pragma unroll
    for (int w = 0; w < 4; ++w) {
        int tt = (dir == 0) ? (t - 3 + w) : (t + 3 - w);
        if (tt >= 0 && tt < SEQ)
            acc += conv_w[c * 4 + w] * xz[((long)(b * SEQ + tt)) * (2 * D_INNER) + c];
    }
    xi[idx] = acc / (1.0f + __expf(-acc)); // silu
}

// ---------------------------------------------------------------------------
// Selective scan, fused with D-skip and silu(z) gating.
// block = 128 threads (one d-chunk), grid = (D_INNER/128, BATCH).
// Per step: stage B,C (16 floats each) in LDS; 16-state recurrence in regs.
// ---------------------------------------------------------------------------
__global__ void scan_kernel(const float* __restrict__ dtb,
                            const float* __restrict__ xi,
                            const float* __restrict__ xdbl,
                            const float* __restrict__ A_log,
                            const float* __restrict__ Dp,
                            const float* __restrict__ xz,
                            float* __restrict__ y, int dir) {
    const int b = blockIdx.y;
    const int d = blockIdx.x * 128 + threadIdx.x;

    float Ac[D_STATE];
#pragma unroll
    for (int s = 0; s < D_STATE; ++s) Ac[s] = -__expf(A_log[d * D_STATE + s]);
    const float Dv = Dp[d];

    float h[D_STATE];
#pragma unroll
    for (int s = 0; s < D_STATE; ++s) h[s] = 0.0f;

    __shared__ float sB[D_STATE];
    __shared__ float sC[D_STATE];

    for (int step = 0; step < SEQ; ++step) {
        int t = (dir == 0) ? step : (SEQ - 1 - step);
        long row = (long)(b * SEQ + t);
        __syncthreads(); // protect previous iteration's readers
        if (threadIdx.x < 32) {
            float v = xdbl[row * 48 + DT_RANK + threadIdx.x]; // B at 16..31, C at 32..47
            if (threadIdx.x < 16) sB[threadIdx.x] = v;
            else                  sC[threadIdx.x - 16] = v;
        }
        // prefetch next step's B/C row into cache (gfx1250 global_prefetch_b8)
        if (threadIdx.x == 0 && step + 1 < SEQ) {
            int tn = (dir == 0) ? (t + 1) : (t - 1);
            __builtin_prefetch(xdbl + (long)(b * SEQ + tn) * 48 + DT_RANK, 0, 0);
        }
        __syncthreads();

        float dtv = dtb[row * D_INNER + d];
        float xv  = xi[row * D_INNER + d];
        float acc = 0.0f;
#pragma unroll
        for (int s = 0; s < D_STATE; ++s) {
            float dA = __expf(dtv * Ac[s]);
            h[s] = dA * h[s] + dtv * sB[s] * xv;
            acc += h[s] * sC[s];
        }
        float zv = xz[row * (2 * D_INNER) + D_INNER + d];
        float g  = zv / (1.0f + __expf(-zv));
        y[row * D_INNER + d] = (acc + Dv * xv) * g;
    }
}

// ---------------------------------------------------------------------------
// Mean-pool over L of (cur_f + cur_r): grid = BATCH blocks x 256 threads
// ---------------------------------------------------------------------------
__global__ void pool_kernel(const float* __restrict__ cf,
                            const float* __restrict__ cr,
                            float* __restrict__ pooled) {
    int b = blockIdx.x;
    int dm = threadIdx.x;
    float s = 0.0f;
    for (int t = 0; t < SEQ; ++t) {
        long row = (long)(b * SEQ + t);
        s += cf[row * D_MODEL + dm] + cr[row * D_MODEL + dm];
    }
    pooled[b * D_MODEL + dm] = s * (1.0f / (float)SEQ);
}

// ---------------------------------------------------------------------------
// Final projection: (16x256) @ (128x256)^T + b. grid = BATCH x 128 threads
// ---------------------------------------------------------------------------
__global__ void proj_kernel(const float* __restrict__ pooled,
                            const float* __restrict__ pw,
                            const float* __restrict__ pb,
                            float* __restrict__ out) {
    int b = blockIdx.x;
    int o = threadIdx.x;
    float s = pb[o];
    const float* wrow = pw + (long)o * D_MODEL;
    const float* prow = pooled + (long)b * D_MODEL;
#pragma unroll 8
    for (int j = 0; j < D_MODEL; ++j) s += prow[j] * wrow[j];
    out[b * 128 + o] = s;
}

// ---------------------------------------------------------------------------
// Host-side orchestration
// Input flattening (setup_inputs dict order, depth-first):
//  0: x | 1..11: emb params | 12+ (dir*4+i)*9 + j: layer params | 84: proj_w | 85: proj_b
// ---------------------------------------------------------------------------
extern "C" void kernel_launch(void* const* d_in, const int* in_sizes, int n_in,
                              void* d_out, int out_size, void* d_ws, size_t ws_size,
                              hipStream_t stream) {
    (void)in_sizes; (void)n_in; (void)out_size; (void)ws_size;
    const float* X = (const float*)d_in[0];
    auto F = [&](int i) { return (const float*)d_in[i]; };

    // workspace carve-out (256B aligned)
    char* ws = (char*)d_ws;
    size_t off = 0;
    auto alloc = [&](size_t bytes) -> char* {
        char* p = ws + off;
        off += (bytes + 255) & ~(size_t)255;
        return p;
    };
    const size_t MD  = (size_t)MTOK * D_MODEL * sizeof(float);
    const size_t MDI = (size_t)MTOK * D_INNER * sizeof(float);

    float*  cur_f  = (float*)alloc(MD);
    float*  cur_r  = (float*)alloc(MD);
    float*  xz     = (float*)alloc(2 * MDI);
    float*  xi     = (float*)alloc(MDI);
    float*  xdbl   = (float*)alloc((size_t)MTOK * 48 * sizeof(float));
    float*  dtb    = (float*)alloc(MDI);
    float*  ybuf   = (float*)alloc(MDI);
    float*  pooled = (float*)alloc((size_t)BATCH * D_MODEL * sizeof(float));
    bf16_t* wb_in  = (bf16_t*)alloc((size_t)2 * D_INNER * D_MODEL * sizeof(bf16_t));
    bf16_t* wb_xp  = (bf16_t*)alloc((size_t)48 * D_INNER * sizeof(bf16_t));
    bf16_t* wb_dt  = (bf16_t*)alloc((size_t)D_INNER * 32 * sizeof(bf16_t)); // padded K=16->32
    bf16_t* wb_out = (bf16_t*)alloc((size_t)D_MODEL * D_INNER * sizeof(bf16_t));

    // Embedder -> both chain buffers
    embed_kernel<<<MTOK, 256, 0, stream>>>(
        X, F(1), F(2), F(3), F(4), F(5), F(6), F(7), F(8), F(9), F(10), F(11),
        cur_f, cur_r);

    for (int dir = 0; dir < 2; ++dir) {
        float* cur = (dir == 0) ? cur_f : cur_r;
        for (int layer = 0; layer < N_LAYERS; ++layer) {
            int base = 12 + (dir * N_LAYERS + layer) * 9;
            const float* in_proj  = F(base + 0);
            const float* conv_w   = F(base + 1);
            const float* conv_b   = F(base + 2);
            const float* x_proj   = F(base + 3);
            const float* dt_w     = F(base + 4);
            const float* dt_b     = F(base + 5);
            const float* A_log    = F(base + 6);
            const float* Dp       = F(base + 7);
            const float* out_proj = F(base + 8);

            // stage weights as bf16 (dt_proj zero-padded to K=32 so GEMM needs no guards)
            {
                int n;
                n = 2 * D_INNER * D_MODEL;
                cvt_bf16_kernel<<<(n + 255) / 256, 256, 0, stream>>>(in_proj, wb_in, n);
                n = 48 * D_INNER;
                cvt_bf16_kernel<<<(n + 255) / 256, 256, 0, stream>>>(x_proj, wb_xp, n);
                n = D_INNER * 32;
                cvt_bf16_pad_kernel<<<(n + 255) / 256, 256, 0, stream>>>(
                    dt_w, wb_dt, D_INNER, DT_RANK, 32);
                n = D_MODEL * D_INNER;
                cvt_bf16_kernel<<<(n + 255) / 256, 256, 0, stream>>>(out_proj, wb_out, n);
            }

            // xz = cur @ in_proj^T   (M=16384, N=1024, K=256)
            wmma_gemm_kernel<2, 4, 0><<<dim3(MTOK / 256, (2 * D_INNER) / 64), 256, 0, stream>>>(
                cur, D_MODEL, wb_in, D_MODEL, xz, 2 * D_INNER,
                MTOK, 2 * D_INNER, D_MODEL, nullptr);

            // depthwise conv + silu -> xi
            conv_silu_kernel<<<(MTOK * D_INNER) / 256, 256, 0, stream>>>(
                xz, conv_w, conv_b, xi, dir);

            // xdbl = xi @ x_proj^T   (N=48, K=512)
            wmma_gemm_kernel<2, 3, 0><<<dim3(MTOK / 256, 1), 256, 0, stream>>>(
                xi, D_INNER, wb_xp, D_INNER, xdbl, 48,
                MTOK, 48, D_INNER, nullptr);

            // dt = softplus(xdbl[:, :16] @ dt_w^T + dt_b)
            // K padded 16->32: A reads xdbl cols 0..31 (in-bounds), W zeros kill k>=16
            wmma_gemm_kernel<2, 4, 2><<<dim3(MTOK / 256, D_INNER / 64), 256, 0, stream>>>(
                xdbl, 48, wb_dt, 32, dtb, D_INNER,
                MTOK, D_INNER, 32, dt_b);

            // selective scan (+ D skip, silu(z) gate) -> ybuf
            scan_kernel<<<dim3(D_INNER / 128, BATCH), 128, 0, stream>>>(
                dtb, xi, xdbl, A_log, Dp, xz, ybuf, dir);

            // cur += ybuf @ out_proj^T   (N=256, K=512), residual fused (MODE 3)
            wmma_gemm_kernel<2, 4, 3><<<dim3(MTOK / 256, D_MODEL / 64), 256, 0, stream>>>(
                ybuf, D_INNER, wb_out, D_INNER, cur, D_MODEL,
                MTOK, D_MODEL, D_INNER, nullptr);
        }
    }

    pool_kernel<<<BATCH, D_MODEL, 0, stream>>>(cur_f, cur_r, pooled);
    proj_kernel<<<BATCH, 128, 0, stream>>>(pooled, F(84), F(85), (float*)d_out);
}
